// AtomAngleProjection_83416854823432
// MI455X (gfx1250) — matrix operand order repaired and checked by reading, not compiled, not run on hardware
//
#include <hip/hip_runtime.h>

// ---------------------------------------------------------------------------
// AtomAngleProjection for MI455X (gfx1250, wave32, WMMA, async-to-LDS).
//
// Shapes: B=64, N_ATOMS=256, D_ATOM=256, T=2048, D_HID=512, D_OUT=256.
// ROWS = B*T = 131072.  ~68.7 GFLOP; bandwidth-bound (~25-30us) once the
// GEMMs run on the bf16 WMMA pipe, so the priority is minimal data movement:
//
//   k0: pack W1/W2 once into their exact LDS image (bf16, [N][K] transposed,
//       padded strides) in d_ws -> per-block staging becomes a raw copy.
//   k1: gather(z, 3 idx) -> bf16 A tile in LDS; weights staged per K-chunk
//       with GLOBAL_LOAD_ASYNC_TO_LDS_B128 (ASYNCcnt); X@W1 via
//       v_wmma_f32_16x16x32_bf16; h stored as bf16 aliased into d_out;
//       deterministic per-block BN partials (no atomics) into d_ws.
//   k2: reduce partials -> scale = gamma*rsqrt(var+eps), shift = beta-mean*scale
//       (b1 cancels exactly under BatchNorm and is skipped).
//   k3: reload h(bf16), affine+ReLU while staging to LDS, @W2 via WMMA, +b2,
//       overwrite the SAME byte range with f32 out (read-before-write per block).
// ---------------------------------------------------------------------------

typedef __attribute__((ext_vector_type(16))) __bf16 v16bf;
typedef __attribute__((ext_vector_type(8)))  float  v8f;

#define ROWS      131072
#define D_ATOM    256
#define D_HID     512
#define D_OUT     256
#define TILE_M    64
#define NBLK      (ROWS / TILE_M)        // 2048 row-tile blocks
#define STRIDE_A1 264                    // 256 + 8 bf16 pad (conflict-free frags)
#define STRIDE_B1 72                     // 64 + 8
#define STRIDE_H  520                    // 512 + 8
#define STRIDE_B3 136                    // 128 + 8
#define STRIDE_CO 260                    // 256 + 4 f32 pad

#define W1_CH  (D_HID * STRIDE_B1)       // 36,864 bf16 per K-chunk image
#define W1_IMG (4 * W1_CH)               // 147,456
#define W2_CH  (D_OUT * STRIDE_B3)       // 34,816
#define W2_IMG (4 * W2_CH)               // 139,264

#define SMEM1 (TILE_M * STRIDE_A1 * 2 + D_HID * STRIDE_B1 * 2)                 // 107,520 B
#define SMEM3 (1024 * 4 + TILE_M * STRIDE_H * 2 + D_OUT * STRIDE_B3 * 2)       // 140,288 B

union FragU { v16bf v; uint4 u[2]; };

__device__ __forceinline__ unsigned short f32_to_bf16(float f) {
  unsigned u = __builtin_bit_cast(unsigned, f);
  u += 0x7FFFu + ((u >> 16) & 1u);               // round-to-nearest-even
  return (unsigned short)(u >> 16);
}
__device__ __forceinline__ float bf16_to_f32(unsigned short h) {
  unsigned u = ((unsigned)h) << 16;
  return __builtin_bit_cast(float, u);
}
__device__ __forceinline__ unsigned pack2(float a, float b) {
  return (unsigned)f32_to_bf16(a) | ((unsigned)f32_to_bf16(b) << 16);
}

// Async copy of `bytes` (multiple of 256 threads * 16B) global -> LDS.
// Uses GLOBAL_LOAD_ASYNC_TO_LDS_B128: per-lane 16B, no VGPR data round-trip,
// tracked by ASYNCcnt. Waits on ASYNCcnt itself; caller must barrier before reading.
__device__ __forceinline__ void async_copy_to_lds(void* lds, const void* gsrc,
                                                  int iters, int tid) {
  unsigned ldsBase = (unsigned)(unsigned long long)lds;   // addr[31:0] = LDS offset
  unsigned long long gBase = (unsigned long long)gsrc;
#pragma unroll
  for (int it = 0; it < 8; ++it) {            // common prefix (both 17 & 18 iters)
    unsigned off = (unsigned)(it * 256 + tid) * 16u;
    unsigned la = ldsBase + off;
    unsigned long long ga = gBase + off;
    asm volatile("global_load_async_to_lds_b128 %0, %1, off"
                 :: "v"(la), "v"(ga) : "memory");
  }
  for (int it = 8; it < iters; ++it) {
    unsigned off = (unsigned)(it * 256 + tid) * 16u;
    unsigned la = ldsBase + off;
    unsigned long long ga = gBase + off;
    asm volatile("global_load_async_to_lds_b128 %0, %1, off"
                 :: "v"(la), "v"(ga) : "memory");
  }
  asm volatile("s_wait_asynccnt 0x0" ::: "memory");
}

// A-frag (16x32 bf16): lanes 0-15 = rows M0..15 (lanes 16-31 same rows).
// Per ISA: half h lane holds K = {8h..8h+7} U {16+8h..16+8h+7} -> 2x ds_read_b128.
__device__ __forceinline__ v16bf load_frag_a(const unsigned short* s, int row0,
                                             int stride, int kbase, int lane) {
  int m = lane & 15, half = lane >> 4;
  const unsigned short* p = s + (size_t)(row0 + m) * stride + kbase + 8 * half;
  FragU f;
  f.u[0] = *(const uint4*)(p);
  f.u[1] = *(const uint4*)(p + 16);
  return f.v;
}
// B-frag (32x16 bf16) from [N][K]-transposed LDS: lane n = lane&15,
// half h lane holds K = {8h..8h+7} U {16+8h..16+8h+7} -> 2x b128.
__device__ __forceinline__ v16bf load_frag_b(const unsigned short* s, int col0,
                                             int stride, int kbase, int lane) {
  int n = lane & 15, half = lane >> 4;
  const unsigned short* p = s + (size_t)(col0 + n) * stride + kbase + 8 * half;
  FragU f;
  f.u[0] = *(const uint4*)(p);
  f.u[1] = *(const uint4*)(p + 16);
  return f.v;
}

__device__ __forceinline__ v8f wmma_bf16(v16bf a, v16bf b, v8f c) {
  return __builtin_amdgcn_wmma_f32_16x16x32_bf16(false, a, false, b,
                                                 (short)0, c, false, false);
}

// ============================ Kernel 0 ======================================
// Pack W1/W2 (f32 row-major [K][N]) into bf16 LDS images: per K-chunk,
// [N][K_chunk] with padded stride; pad entries zero-filled.
__global__ void __launch_bounds__(256)
k0_pack_weights(const float* __restrict__ W1, const float* __restrict__ W2,
                unsigned short* __restrict__ w1i, unsigned short* __restrict__ w2i) {
  int i = blockIdx.x * 256 + threadIdx.x;
  if (i < W1_IMG) {
    int kc = i / W1_CH, r = i % W1_CH;
    int n = r / STRIDE_B1, k = r % STRIDE_B1;
    unsigned short v = 0;
    if (k < 64) v = f32_to_bf16(W1[(size_t)(kc * 64 + k) * D_HID + n]);
    w1i[i] = v;
  }
  if (i < W2_IMG) {
    int kc = i / W2_CH, r = i % W2_CH;
    int n = r / STRIDE_B3, k = r % STRIDE_B3;
    unsigned short v = 0;
    if (k < 128) v = f32_to_bf16(W2[(size_t)(kc * 128 + k) * D_OUT + n]);
    w2i[i] = v;
  }
}

// ============================ Kernel 1 ======================================
__global__ void __launch_bounds__(256)
k1_gather_gemm1(const float* __restrict__ z, const long long* __restrict__ tbl,
                const unsigned short* __restrict__ w1img,
                unsigned short* __restrict__ hbuf,
                float* __restrict__ wsSum, float* __restrict__ wsSq) {
  extern __shared__ char smem[];
  unsigned short* lA = (unsigned short*)smem;                            // [64][264]
  unsigned short* lB = (unsigned short*)(smem + TILE_M * STRIDE_A1 * 2); // [512][72]

  const int tid = threadIdx.x, lane = tid & 31, wave = tid >> 5;
  const int r0 = blockIdx.x * TILE_M;
  const int wm = wave >> 2;     // 0..1 : 32-row slab
  const int wn = wave & 3;      // 0..3 : 128-col slab

  // ---- Stage A: gather 3 z-rows per angle, sum, f32->bf16, into LDS -------
  for (int i = 0; i < 8; ++i) {
    int row = wave * 8 + i;
    int g = r0 + row;
    int b = g >> 11;                               // g / T_ANGLES
    const long long* tp = tbl + (size_t)g * 3;     // wave-uniform -> SMEM loads
    int i0 = (int)tp[0], i1 = (int)tp[1], i2 = (int)tp[2];
    const float* z0 = z + ((size_t)b * 256 + i0) * D_ATOM + lane * 8;
    const float* z1 = z + ((size_t)b * 256 + i1) * D_ATOM + lane * 8;
    const float* z2 = z + ((size_t)b * 256 + i2) * D_ATOM + lane * 8;
    float4 a0 = *(const float4*)(z0), a1 = *(const float4*)(z0 + 4);
    float4 c0 = *(const float4*)(z1), c1 = *(const float4*)(z1 + 4);
    float4 d0 = *(const float4*)(z2), d1 = *(const float4*)(z2 + 4);
    float s0 = a0.x + c0.x + d0.x, s1 = a0.y + c0.y + d0.y;
    float s2 = a0.z + c0.z + d0.z, s3 = a0.w + c0.w + d0.w;
    float s4 = a1.x + c1.x + d1.x, s5 = a1.y + c1.y + d1.y;
    float s6 = a1.z + c1.z + d1.z, s7 = a1.w + c1.w + d1.w;
    uint4 pk;
    pk.x = pack2(s0, s1); pk.y = pack2(s2, s3);
    pk.z = pack2(s4, s5); pk.w = pack2(s6, s7);
    *(uint4*)(lA + (size_t)row * STRIDE_A1 + lane * 8) = pk;
  }

  v8f acc[2][8];
  const v8f vzero = {0.f, 0.f, 0.f, 0.f, 0.f, 0.f, 0.f, 0.f};
#pragma unroll
  for (int m = 0; m < 2; ++m)
#pragma unroll
    for (int n = 0; n < 8; ++n) acc[m][n] = vzero;

  // ---- K loop: 4 chunks of 64; preformed W1 image async-copied to LDS ----
  for (int kc = 0; kc < 4; ++kc) {
    __syncthreads();                    // prior frag reads done / A ready
    async_copy_to_lds(lB, w1img + (size_t)kc * W1_CH, 18, tid);   // 73,728 B
    __syncthreads();
#pragma unroll
    for (int k32 = 0; k32 < 2; ++k32) {
      int kb = k32 * 32;
      v16bf fa0 = load_frag_a(lA, wm * 32,      STRIDE_A1, kc * 64 + kb, lane);
      v16bf fa1 = load_frag_a(lA, wm * 32 + 16, STRIDE_A1, kc * 64 + kb, lane);
#pragma unroll
      for (int nt = 0; nt < 8; ++nt) {
        v16bf fb = load_frag_b(lB, wn * 128 + nt * 16, STRIDE_B1, kb, lane);
        acc[0][nt] = wmma_bf16(fa0, fb, acc[0][nt]);
        acc[1][nt] = wmma_bf16(fa1, fb, acc[1][nt]);
      }
    }
  }

  // ---- Epilogue: h -> LDS (bf16), BN partials, coalesced store to hbuf ----
  __syncthreads();
  unsigned short* hs = (unsigned short*)smem;    // reuse as [64][520]
#pragma unroll
  for (int mt = 0; mt < 2; ++mt)
#pragma unroll
    for (int nt = 0; nt < 8; ++nt) {
      int row0 = wm * 32 + mt * 16 + 8 * (lane >> 4);
      int col  = wn * 128 + nt * 16 + (lane & 15);
#pragma unroll
      for (int r = 0; r < 8; ++r)
        hs[(size_t)(row0 + r) * STRIDE_H + col] = f32_to_bf16(acc[mt][nt][r]);
    }
  __syncthreads();

  for (int j = tid; j < D_HID; j += 256) {       // per-feature partial stats
    float s = 0.f, q = 0.f;
    for (int i = 0; i < TILE_M; ++i) {
      float v = bf16_to_f32(hs[(size_t)i * STRIDE_H + j]);
      s += v; q += v * v;
    }
    wsSum[(size_t)blockIdx.x * D_HID + j] = s;
    wsSq [(size_t)blockIdx.x * D_HID + j] = q;
  }
  for (int it = 0; it < 16; ++it) {              // h tile -> global (bf16)
    int idx = it * 256 + tid;
    int off = idx * 8;
    int row = off >> 9, j = off & 511;
    uint4 v = *(const uint4*)(hs + (size_t)row * STRIDE_H + j);
    *(uint4*)(hbuf + ((size_t)(r0 + row) * D_HID + j)) = v;
  }
}

// ============================ Kernel 2 ======================================
__global__ void __launch_bounds__(512)
k2_stats(const float* __restrict__ wsSum, const float* __restrict__ wsSq,
         const float* __restrict__ gamma, const float* __restrict__ beta,
         float* __restrict__ scl, float* __restrict__ shf) {
  int j = threadIdx.x;                           // 512 features
  float s = 0.f, q = 0.f;
  for (int b = 0; b < NBLK; ++b) {
    s += wsSum[(size_t)b * D_HID + j];
    q += wsSq [(size_t)b * D_HID + j];
  }
  const float invN = 1.0f / (float)ROWS;
  float mean = s * invN;
  float var  = fmaxf(q * invN - mean * mean, 0.f);   // biased var (ddof=0)
  float sc   = gamma[j] * rsqrtf(var + 1e-5f);
  scl[j] = sc;
  shf[j] = beta[j] - mean * sc;                  // b1 cancels under BN exactly
}

// ============================ Kernel 3 ======================================
__global__ void __launch_bounds__(256)
k3_bn_relu_gemm2(const unsigned short* hbuf,                 // aliases `out`!
                 const float* __restrict__ scl, const float* __restrict__ shf,
                 const unsigned short* __restrict__ w2img,
                 const float* __restrict__ b2, float* out) {
  extern __shared__ char smem[];
  float* sS = (float*)smem;                                   // [512]
  float* sB = sS + 512;                                       // [512]
  unsigned short* ha = (unsigned short*)(sB + 512);           // [64][520] bf16
  unsigned short* wb = ha + TILE_M * STRIDE_H;                // [256][136] bf16
  float* co = (float*)ha;                                     // reuse: [64][260] f32

  const int tid = threadIdx.x, lane = tid & 31, wave = tid >> 5;
  const int r0 = blockIdx.x * TILE_M;
  const int wm = wave >> 2;    // 0..1 : 32-row slab
  const int wn = wave & 3;     // 0..3 : 64-col slab

  for (int j = tid; j < D_HID; j += 256) { sS[j] = scl[j]; sB[j] = shf[j]; }
  __syncthreads();

  // ---- Stage A: reload bf16 h, apply BN affine + ReLU, into LDS ----------
  for (int it = 0; it < 16; ++it) {
    int idx = it * 256 + tid;
    int off = idx * 8;
    int row = off >> 9, j = off & 511;
    uint4 v = *(const uint4*)(hbuf + ((size_t)(r0 + row) * D_HID + j));
    const unsigned short* hp = (const unsigned short*)&v;
    float y[8];
#pragma unroll
    for (int e = 0; e < 8; ++e) {
      float x = bf16_to_f32(hp[e]) * sS[j + e] + sB[j + e];
      y[e] = x > 0.f ? x : 0.f;
    }
    uint4 pk;
    pk.x = pack2(y[0], y[1]); pk.y = pack2(y[2], y[3]);
    pk.z = pack2(y[4], y[5]); pk.w = pack2(y[6], y[7]);
    *(uint4*)(ha + (size_t)row * STRIDE_H + j) = pk;
  }

  v8f acc[2][4];
  const v8f vzero = {0.f, 0.f, 0.f, 0.f, 0.f, 0.f, 0.f, 0.f};
#pragma unroll
  for (int m = 0; m < 2; ++m)
#pragma unroll
    for (int n = 0; n < 4; ++n) acc[m][n] = vzero;

  // ---- K loop: 4 chunks of 128; preformed W2 image async-copied to LDS ---
  for (int kc = 0; kc < 4; ++kc) {
    __syncthreads();                 // covers initial A staging too
    async_copy_to_lds(wb, w2img + (size_t)kc * W2_CH, 17, tid);   // 69,632 B
    __syncthreads();
#pragma unroll
    for (int k32 = 0; k32 < 4; ++k32) {
      int kbA = kc * 128 + k32 * 32;
      int kbB = k32 * 32;
      v16bf fa0 = load_frag_a(ha, wm * 32,      STRIDE_H, kbA, lane);
      v16bf fa1 = load_frag_a(ha, wm * 32 + 16, STRIDE_H, kbA, lane);
#pragma unroll
      for (int nt = 0; nt < 4; ++nt) {
        v16bf fb = load_frag_b(wb, wn * 64 + nt * 16, STRIDE_B3, kbB, lane);
        acc[0][nt] = wmma_bf16(fa0, fb, acc[0][nt]);
        acc[1][nt] = wmma_bf16(fa1, fb, acc[1][nt]);
      }
    }
  }

  // ---- Epilogue: acc -> LDS (f32), +b2, coalesced f32 stores -------------
  __syncthreads();
#pragma unroll
  for (int mt = 0; mt < 2; ++mt)
#pragma unroll
    for (int nt = 0; nt < 4; ++nt) {
      int row0 = wm * 32 + mt * 16 + 8 * (lane >> 4);
      int col  = wn * 64 + nt * 16 + (lane & 15);
#pragma unroll
      for (int r = 0; r < 8; ++r)
        co[(size_t)(row0 + r) * STRIDE_CO + col] = acc[mt][nt][r];
    }
  __syncthreads();
  for (int it = 0; it < 16; ++it) {
    int idx = it * 256 + tid;
    int off = idx * 4;
    int row = off >> 8, j = off & 255;
    float4 v  = *(const float4*)(co + (size_t)row * STRIDE_CO + j);
    float4 bb = *(const float4*)(b2 + j);
    v.x += bb.x; v.y += bb.y; v.z += bb.z; v.w += bb.w;
    *(float4*)(out + ((size_t)(r0 + row) * D_OUT + j)) = v;   // same bytes h lived in
  }
}

// ============================ Launch ========================================
extern "C" void kernel_launch(void* const* d_in, const int* in_sizes, int n_in,
                              void* d_out, int out_size, void* d_ws, size_t ws_size,
                              hipStream_t stream) {
  const float*     z     = (const float*)d_in[0];
  const long long* tbl   = (const long long*)d_in[1];
  const float*     W1    = (const float*)d_in[2];
  // d_in[3] = b1 : mathematically cancelled by BatchNorm's mean subtraction.
  const float*     gamma = (const float*)d_in[4];
  const float*     beta  = (const float*)d_in[5];
  const float*     W2    = (const float*)d_in[6];
  const float*     b2    = (const float*)d_in[7];
  float* out = (float*)d_out;

  // h (bf16, ROWS x 512 = 128 MiB) aliases d_out (f32, ROWS x 256 = 128 MiB).
  unsigned short* hbuf = (unsigned short*)d_out;

  float* ws      = (float*)d_ws;
  float* wsSum   = ws;                               // [2048][512]
  float* wsSq    = ws + (size_t)NBLK * D_HID;        // [2048][512]
  float* wsScale = ws + (size_t)2 * NBLK * D_HID;    // [512]
  float* wsShift = wsScale + D_HID;                  // [512]
  unsigned short* w1img = (unsigned short*)(wsShift + D_HID);  // 147,456 bf16
  unsigned short* w2img = w1img + W1_IMG;                      // 139,264 bf16
                                                     // total ws ~8.6 MiB

  k0_pack_weights<<<(W1_IMG + 255) / 256, 256, 0, stream>>>(W1, W2, w1img, w2img);
  k1_gather_gemm1<<<NBLK, 256, SMEM1, stream>>>(z, tbl, w1img, hbuf, wsSum, wsSq);
  k2_stats<<<1, 512, 0, stream>>>(wsSum, wsSq, gamma, beta, wsScale, wsShift);
  k3_bn_relu_gemm2<<<NBLK, 256, SMEM3, stream>>>(hbuf, wsScale, wsShift, w2img, b2, out);
}
